// GATModel_35777077575717
// MI455X (gfx1250) — compile-verified, matrix operand order checked by reading, not run on hardware
//
#include <hip/hip_runtime.h>
#include <math.h>

typedef __attribute__((ext_vector_type(16))) _Float16 v16h;
typedef __attribute__((ext_vector_type(8)))  _Float16 v8h;
typedef __attribute__((ext_vector_type(8)))  float    v8f;

#define B_ 16
#define S_ 1024
#define D_ 512
#define O_ 512
#define THRESH_ 0.9f
#define NEG_SLOPE_ 0.2f

// Load a 16x32 f16 A-layout fragment (also used for B = row-major rows of B^T).
// p must already point at: base + row(lane&15)*ld + (lane>>4)*8 + k0.
// Layout (ISA 7.12.2, 16-bit A 16x32): half h lanes hold K = h*8..h*8+7 (VGPR0-3)
// and K = 16+h*8..16+h*8+7 (VGPR4-7) -> two contiguous 8-element (16B) chunks.
static __device__ __forceinline__ v16h load_frag(const _Float16* p) {
    v8h lo = *(const v8h*)(p);
    v8h hi = *(const v8h*)(p + 16);
    v16h r;
#pragma unroll
    for (int i = 0; i < 8; ++i) { r[i] = lo[i]; r[i + 8] = hi[i]; }
    return r;
}

static __device__ __forceinline__ v8f wmma16(v16h a, v16h b, v8f c) {
    return __builtin_amdgcn_wmma_f32_16x16x32_f16(
        /*neg_a=*/false, a, /*neg_b=*/false, b,
        /*c_mod=*/(short)0, c, /*reuse_a=*/false, /*reuse_b=*/false);
}

// ---------------------------------------------------------------------------
// Kernel 1: per-row L2 norm; emit x (f16) and x/||x|| (f16). 1 wave per row.
// ---------------------------------------------------------------------------
__global__ __launch_bounds__(256) void k_norm(const float* __restrict__ x,
                                              _Float16* __restrict__ xh,
                                              _Float16* __restrict__ xn) {
    int row  = blockIdx.x * 8 + (threadIdx.x >> 5);
    int lane = threadIdx.x & 31;
    const float* xr = x + (size_t)row * D_ + lane * 16;
    float v[16];
    float ss = 0.f;
#pragma unroll
    for (int i = 0; i < 16; ++i) { v[i] = xr[i]; ss += v[i] * v[i]; }
#pragma unroll
    for (int off = 16; off > 0; off >>= 1) ss += __shfl_xor(ss, off, 32);
    float inv = 1.0f / fmaxf(sqrtf(ss), 1e-12f);

    _Float16* xo = xh + (size_t)row * D_ + lane * 16;
    _Float16* no = xn + (size_t)row * D_ + lane * 16;
    v8h h0, h1, n0, n1;
#pragma unroll
    for (int i = 0; i < 8; ++i) {
        h0[i] = (_Float16)v[i];
        h1[i] = (_Float16)v[i + 8];
        n0[i] = (_Float16)(v[i] * inv);
        n1[i] = (_Float16)(v[i + 8] * inv);
    }
    *(v8h*)(xo)     = h0;
    *(v8h*)(xo + 8) = h1;
    *(v8h*)(no)     = n0;
    *(v8h*)(no + 8) = n1;
}

// ---------------------------------------------------------------------------
// Kernel 2: Wt[o][d] = (f16) W[d][o]  (so B-fragments load row-major)
// ---------------------------------------------------------------------------
__global__ __launch_bounds__(256) void k_wt(const float* __restrict__ W,
                                            _Float16* __restrict__ Wt) {
    int idx = blockIdx.x * 256 + threadIdx.x;
    int o = idx >> 9;
    int d = idx & 511;
    Wt[(size_t)o * D_ + d] = (_Float16)W[(size_t)d * O_ + o];
}

// ---------------------------------------------------------------------------
// Kernel 3: h = x @ W via WMMA; store hT f16 [b][o][s].
// Block = 8 waves; M-tile 16 rows; wave w owns 64 output cols.
// K loop fully unrolled with parity double-buffering (no register rotation).
// ---------------------------------------------------------------------------
__global__ __launch_bounds__(256) void k_gemm_h(const _Float16* __restrict__ xh,
                                                const _Float16* __restrict__ Wt,
                                                _Float16* __restrict__ hT) {
    int b = blockIdx.y;
    int m0 = blockIdx.x * 16;
    int wave = threadIdx.x >> 5, lane = threadIdx.x & 31;
    int lm = lane & 15, half = lane >> 4;
    int n0 = wave * 64;

    const _Float16* Ab = xh + (size_t)(b * S_ + m0 + lm) * D_ + half * 8;
    const _Float16* Bbase[4];
#pragma unroll
    for (int j = 0; j < 4; ++j)
        Bbase[j] = Wt + (size_t)(n0 + j * 16 + lm) * D_ + half * 8;

    constexpr int NK = D_ / 32;  // 16
    v8f acc[4] = {};
    v16h a[2];
    v16h bb[2][4];
    a[0] = load_frag(Ab);
#pragma unroll
    for (int j = 0; j < 4; ++j) bb[0][j] = load_frag(Bbase[j]);

#pragma unroll
    for (int ks = 0; ks < NK; ++ks) {
        const int cur = ks & 1, nxt = cur ^ 1;
        if (ks + 1 < NK) {
            a[nxt] = load_frag(Ab + (ks + 1) * 32);
#pragma unroll
            for (int j = 0; j < 4; ++j)
                bb[nxt][j] = load_frag(Bbase[j] + (ks + 1) * 32);
        }
#pragma unroll
        for (int j = 0; j < 4; ++j)
            acc[j] = wmma16(a[cur], bb[cur][j], acc[j]);
    }

#pragma unroll
    for (int j = 0; j < 4; ++j) {
        int n = n0 + j * 16 + lm;
        _Float16* dp = hT + (size_t)b * O_ * S_ + (size_t)n * S_ + m0;
#pragma unroll
        for (int r = 0; r < 8; ++r) {
            dp[r + 8 * half] = (_Float16)acc[j][r];
        }
    }
}

// ---------------------------------------------------------------------------
// Kernel 4: a_src[b,s] = h[b,s]·att_src ; a_dst likewise (coalesced over s).
// ---------------------------------------------------------------------------
__global__ __launch_bounds__(256) void k_att(const _Float16* __restrict__ hT,
                                             const float* __restrict__ att_src,
                                             const float* __restrict__ att_dst,
                                             float* __restrict__ asrc,
                                             float* __restrict__ adst) {
    int b = blockIdx.y;
    int s = blockIdx.x * 256 + threadIdx.x;
    const _Float16* hp = hT + (size_t)b * O_ * S_ + s;
    float as = 0.f, ad = 0.f;
    for (int o = 0; o < O_; ++o) {
        float v = (float)hp[(size_t)o * S_];
        as += v * att_src[o];
        ad += v * att_dst[o];
    }
    asrc[b * S_ + s] = as;
    adst[b * S_ + s] = ad;
}

// ---------------------------------------------------------------------------
// Kernel 5: fused sim(WMMA) -> mask+leakyrelu -> softmax -> attn@h (WMMA)
// One block per (b, 16 dst rows). 8 waves. Both WMMA loops parity-pipelined.
// ---------------------------------------------------------------------------
__global__ __launch_bounds__(256) void k_attn(const _Float16* __restrict__ xn,
                                              const _Float16* __restrict__ hT,
                                              const float* __restrict__ asrc,
                                              const float* __restrict__ adst,
                                              const float* __restrict__ bias,
                                              float* __restrict__ out) {
    __shared__ _Float16 xnd[16 * 512];    // 16 KB: normalized dst rows
    __shared__ float    ebuf[16 * 1024];  // 64 KB: masked logits
    __shared__ _Float16 pbuf[16 * 1024];  // 32 KB: probabilities (f16)
    __shared__ float    red[16 * 16];
    __shared__ float    mrow[16];
    __shared__ float    srow[16];
    __shared__ float    adl[16];

    int b = blockIdx.y;
    int d0 = blockIdx.x * 16;
    int t = threadIdx.x;
    int wave = t >> 5, lane = t & 31, lm = lane & 15, half = lane >> 4;

    // stage 16 normalized dst rows (8192 halfs) into LDS
    {
        const v8h* src8 = (const v8h*)(xn + (size_t)(b * S_ + d0) * D_);
        v8h* dst8 = (v8h*)xnd;
#pragma unroll
        for (int i = 0; i < 4; ++i) dst8[t + 256 * i] = src8[t + 256 * i];
    }
    if (t < 16) adl[t] = adst[b * S_ + d0 + t];
    __syncthreads();

    // ---- pass 1: similarity (16 dst x 16 src, K=512) + masked leaky-relu ----
    const _Float16* Ab = xnd + lm * D_ + half * 8;
    constexpr int NK1 = D_ / 32;  // 16
    for (int st = wave; st < 64; st += 8) {
        const _Float16* Bb =
            xn + (size_t)(b * S_ + st * 16 + lm) * D_ + half * 8;
        v8f c = {};
        v16h a[2], bbv[2];
        a[0]   = load_frag(Ab);
        bbv[0] = load_frag(Bb);
#pragma unroll
        for (int ks = 0; ks < NK1; ++ks) {
            const int cur = ks & 1, nxt = cur ^ 1;
            if (ks + 1 < NK1) {
                a[nxt]   = load_frag(Ab + (ks + 1) * 32);
                bbv[nxt] = load_frag(Bb + (ks + 1) * 32);
            }
            c = wmma16(a[cur], bbv[cur], c);
        }

        int src = st * 16 + lm;
        float asv = asrc[b * S_ + src];
#pragma unroll
        for (int r = 0; r < 8; ++r) {
            int m = r + 8 * half;
            int dst = d0 + m;
            float sim = c[r];
            bool edge = ((sim > THRESH_) && (src < dst)) || (src == dst);
            float ev = -1e30f;
            if (edge) {
                float z = adl[m] + asv;
                ev = (z > 0.f) ? z : NEG_SLOPE_ * z;
            }
            ebuf[m * S_ + src] = ev;
        }
    }
    __syncthreads();

    // ---- softmax over src (row = dst), 16 threads x 64 cols each ----
    int row = t >> 4, seg = t & 15;
    {
        float mx = -3e38f;
        const float* er = ebuf + row * S_ + seg * 64;
        for (int i = 0; i < 64; ++i) mx = fmaxf(mx, er[i]);
        red[row * 16 + seg] = mx;
    }
    __syncthreads();
    if (t < 16) {
        float mx = -3e38f;
        for (int i = 0; i < 16; ++i) mx = fmaxf(mx, red[t * 16 + i]);
        mrow[t] = mx;
    }
    __syncthreads();
    {
        float sm = 0.f, mx = mrow[row];
        const float* er = ebuf + row * S_ + seg * 64;
        for (int i = 0; i < 64; ++i) sm += __expf(er[i] - mx);
        red[row * 16 + seg] = sm;
    }
    __syncthreads();
    if (t < 16) {
        float sm = 0.f;
        for (int i = 0; i < 16; ++i) sm += red[t * 16 + i];
        srow[t] = 1.0f / sm;
    }
    __syncthreads();
    {
        float mx = mrow[row], inv = srow[row];
        const float* er = ebuf + row * S_ + seg * 64;
        _Float16* pr = pbuf + row * S_ + seg * 64;
        for (int i = 0; i < 64; ++i)
            pr[i] = (_Float16)(__expf(er[i] - mx) * inv);
    }
    __syncthreads();

    // ---- pass 2: out(16 x 512) = P(16 x 1024) @ h(1024 x 512) via WMMA ----
    const _Float16* Pb = pbuf + lm * S_ + half * 8;
    int n0 = wave * 64;
    const _Float16* Bbase[4];
#pragma unroll
    for (int j = 0; j < 4; ++j)
        Bbase[j] = hT + (size_t)b * O_ * S_ +
                   (size_t)(n0 + j * 16 + lm) * S_ + half * 8;

    constexpr int NK2 = S_ / 32;  // 32
    v8f acc[4] = {};
    v16h a2[2];
    v16h b2[2][4];
    a2[0] = load_frag(Pb);
#pragma unroll
    for (int j = 0; j < 4; ++j) b2[0][j] = load_frag(Bbase[j]);

#pragma unroll
    for (int ks = 0; ks < NK2; ++ks) {
        const int cur = ks & 1, nxt = cur ^ 1;
        if (ks + 1 < NK2) {
            a2[nxt] = load_frag(Pb + (ks + 1) * 32);
#pragma unroll
            for (int j = 0; j < 4; ++j)
                b2[nxt][j] = load_frag(Bbase[j] + (ks + 1) * 32);
        }
#pragma unroll
        for (int j = 0; j < 4; ++j)
            acc[j] = wmma16(a2[cur], b2[cur][j], acc[j]);
    }

#pragma unroll
    for (int j = 0; j < 4; ++j) {
        int n = n0 + j * 16 + lm;
        float bv = bias[n];
#pragma unroll
        for (int r = 0; r < 8; ++r) {
            int m = r + 8 * half;
            float v = acc[j][r] + bv;
            out[((size_t)b * S_ + d0 + m) * O_ + n] = v > 0.f ? v : 0.f;
        }
    }
}

extern "C" void kernel_launch(void* const* d_in, const int* in_sizes, int n_in,
                              void* d_out, int out_size, void* d_ws, size_t ws_size,
                              hipStream_t stream) {
    const float* x       = (const float*)d_in[0];
    const float* W       = (const float*)d_in[1];
    const float* att_src = (const float*)d_in[2];
    const float* att_dst = (const float*)d_in[3];
    const float* bias    = (const float*)d_in[4];
    float* out = (float*)d_out;

    char* ws = (char*)d_ws;
    size_t off = 0;
    _Float16* xh = (_Float16*)(ws + off); off += (size_t)B_ * S_ * D_ * 2;  // 16 MB
    _Float16* xn = (_Float16*)(ws + off); off += (size_t)B_ * S_ * D_ * 2;  // 16 MB
    _Float16* hT = (_Float16*)(ws + off); off += (size_t)B_ * O_ * S_ * 2;  // 16 MB
    _Float16* Wt = (_Float16*)(ws + off); off += (size_t)D_ * O_ * 2;       // 512 KB
    float* asrc  = (float*)(ws + off);    off += (size_t)B_ * S_ * 4;       // 64 KB
    float* adst  = (float*)(ws + off);    off += (size_t)B_ * S_ * 4;       // 64 KB

    k_norm  <<<dim3(B_ * S_ / 8), 256, 0, stream>>>(x, xh, xn);
    k_wt    <<<dim3(D_ * O_ / 256), 256, 0, stream>>>(W, Wt);
    k_gemm_h<<<dim3(S_ / 16, B_), 256, 0, stream>>>(xh, Wt, hT);
    k_att   <<<dim3(S_ / 256, B_), 256, 0, stream>>>(hT, att_src, att_dst, asrc, adst);
    k_attn  <<<dim3(S_ / 16, B_), 256, 0, stream>>>(xn, hT, asrc, adst, bias, out);
}
